// CriticNetwork_35167192219768
// MI455X (gfx1250) — matrix operand order repaired and checked by reading, not compiled, not run on hardware
//
#include <hip/hip_runtime.h>
#include <hip/hip_bf16.h>

typedef __attribute__((ext_vector_type(16))) __bf16 v16bf;
typedef __attribute__((ext_vector_type(8)))  float  v8f;

#define SD 128   // STATE_DIM
#define H1 96
#define H2 64
#define WSW_ELEMS (4 * 6 * 32 * 16)   // kb * tiles * lanes * elems = 12288 bf16

// ---------------------------------------------------------------------------
// K1: init deg=1 (self-loop) and zero the 96-float pooled accumulator
// ---------------------------------------------------------------------------
__global__ void deg_init_kernel(float* __restrict__ deg, float* __restrict__ g, int N) {
    int i = blockIdx.x * blockDim.x + threadIdx.x;
    if (i < N) deg[i] = 1.0f;
    if (i < H1) g[i] = 0.0f;
}

// ---------------------------------------------------------------------------
// K2: deg[dst] += 1 per edge
// ---------------------------------------------------------------------------
__global__ void deg_count_kernel(const long long* __restrict__ dst, float* __restrict__ deg, int E) {
    int i = blockIdx.x * blockDim.x + threadIdx.x;
    if (i < E) atomicAdd(&deg[(int)dst[i]], 1.0f);
}

// ---------------------------------------------------------------------------
// K3: dinv = rsqrt(deg)  (deg >= 1 always, matches reference's where/max)
// ---------------------------------------------------------------------------
__global__ void deg_rsqrt_kernel(float* __restrict__ deg, int N) {
    int i = blockIdx.x * blockDim.x + threadIdx.x;
    if (i < N) deg[i] = rsqrtf(deg[i]);
}

// ---------------------------------------------------------------------------
// K4a: pre-swizzle W1 into WMMA B-fragment layout (bf16), one time, 24KB.
//      Fragment element e of lane l, tile t, K-block kb:
//        W1[kb*32 + (l>>4)*16 + e][t*16 + (l&15)]
//      stored at wsw[((kb*6 + t)*32 + l)*16 + e]  (32B contiguous per lane)
// ---------------------------------------------------------------------------
__global__ void w_swizzle_kernel(const float* __restrict__ W1, __bf16* __restrict__ wsw) {
    int idx = blockIdx.x * blockDim.x + threadIdx.x;
    if (idx < WSW_ELEMS) {
        const int e    = idx & 15;
        const int lane = (idx >> 4) & 31;
        const int t    = (idx >> 9) % 6;
        const int kb   = idx / (16 * 32 * 6);
        const int k    = kb * 32 + (lane >> 4) * 16 + e;
        const int n    = t * 16 + (lane & 15);
        wsw[idx] = (__bf16)W1[k * H1 + n];
    }
}

// ---------------------------------------------------------------------------
// K4b: hs = (x @ W1) * dinv[row]  via bf16 WMMA, fp32 accumulate.
//      Also writes S = hs (folds the self-loop contribution).
//      One wave computes a 16-row x 96-col stripe (6 tiles); block = 8 waves.
//      Per K-step: stage ALL 6 B fragments (48 VGPRs) + the A fragment, THEN
//      issue the 6 WMMAs — lets the compiler clause the 12 b128 loads and
//      drain them with staggered loadcnt waits instead of load->wait0->wmma.
// ---------------------------------------------------------------------------
__global__ __launch_bounds__(256) void gemm_scale_kernel(
    const float* __restrict__ x, const __bf16* __restrict__ wsw,
    const float* __restrict__ dinv,
    float* __restrict__ hs, float* __restrict__ S, int N)
{
    const int lane = threadIdx.x & 31;
    const int wave = threadIdx.x >> 5;
    const int hf   = lane >> 4;      // half-wave: 0 or 1
    const int lr   = lane & 15;
    const int m0   = blockIdx.x * 128 + wave * 16;

    const v16bf* __restrict__ wv = (const v16bf*)wsw;

    v8f acc[6] = {};                 // 6 tiles of 16x16 f32 accum (48 VGPRs)

    int arow = m0 + lr; if (arow >= N) arow = N - 1;   // clamp, keep EXEC full
    const float* xrow = x + (size_t)arow * SD;

    #pragma unroll
    for (int kb = 0; kb < 4; ++kb) { // K = 128 in 4 steps of 32
        // ---- stage B fragments for all 6 tiles (one 12-load clause) ----
        v16bf bf[6];
        #pragma unroll
        for (int t = 0; t < 6; ++t)
            bf[t] = wv[(kb * 6 + t) * 32 + lane];

        // ---- A fragment: K-chunk [kb*32+hf*16, +16) ----
        const float* xr = xrow + kb * 32 + hf * 16;
        v16bf a;
        #pragma unroll
        for (int e = 0; e < 8; ++e) a[e]     = (__bf16)xr[e];
        #pragma unroll
        for (int e = 0; e < 8; ++e) a[8 + e] = (__bf16)xr[8 + e];

        // ---- 6 back-to-back WMMAs ----
        #pragma unroll
        for (int t = 0; t < 6; ++t)
            acc[t] = __builtin_amdgcn_wmma_f32_16x16x32_bf16(
                false, a, false, bf[t], (short)0, acc[t], false, false);
    }

    // ---- epilogue: scale by dinv[row], store hs and S (=hs for self-loop) ----
    #pragma unroll
    for (int i = 0; i < 8; ++i) {
        const int row = m0 + hf * 8 + i;   // C layout: M = 8*(lane/16)+vgpr, Nc = lane%16
        if (row < N) {
            const float d = dinv[row];
            const size_t base = (size_t)row * H1;
            #pragma unroll
            for (int t = 0; t < 6; ++t) {
                const float v = acc[t][i] * d;
                hs[base + t * 16 + lr] = v;
                S [base + t * 16 + lr] = v;
            }
        }
    }
}

// ---------------------------------------------------------------------------
// K5: per edge: S[dst][:] += hs[src][:]   (one wave32 per edge, 3 iters of 32)
// ---------------------------------------------------------------------------
__global__ __launch_bounds__(256) void edge_scatter_kernel(
    const long long* __restrict__ src, const long long* __restrict__ dst,
    const float* __restrict__ hs, float* __restrict__ S, int E)
{
    const int eid  = (int)((blockIdx.x * blockDim.x + threadIdx.x) >> 5);
    const int lane = threadIdx.x & 31;
    if (eid >= E) return;
    const int s = (int)src[eid];
    const int d = (int)dst[eid];
    const float* hr = hs + (size_t)s * H1;
    float*       Sr = S  + (size_t)d * H1;
    #pragma unroll
    for (int f = lane; f < H1; f += 32)
        atomicAdd(&Sr[f], hr[f]);
}

// ---------------------------------------------------------------------------
// K6: g[f] = sum_v relu(dinv[v]*S[v][f] + b1[f])  (LDS partials, ds atomics)
// ---------------------------------------------------------------------------
__global__ __launch_bounds__(256) void pool_kernel(
    const float* __restrict__ S, const float* __restrict__ dinv,
    const float* __restrict__ b1, float* __restrict__ g, int N)
{
    __shared__ float gp[H1];
    for (int i = threadIdx.x; i < H1; i += blockDim.x) gp[i] = 0.0f;
    __syncthreads();

    const int NPB = 512;                      // nodes per block
    const int v0  = blockIdx.x * NPB;
    for (int e = threadIdx.x; e < NPB * H1; e += blockDim.x) {
        const int v = v0 + e / H1;
        if (v < N) {
            const int f = e % H1;
            const float val = dinv[v] * S[(size_t)v * H1 + f] + b1[f];
            if (val > 0.0f) atomicAdd(&gp[f], val);
        }
    }
    __syncthreads();
    for (int f = threadIdx.x; f < H1; f += blockDim.x)
        atomicAdd(&g[f], gp[f]);
}

// ---------------------------------------------------------------------------
// K7: tiny MLP head, single block: 96 -> relu 96 -> relu 64 -> 1
// ---------------------------------------------------------------------------
__global__ __launch_bounds__(128) void mlp_kernel(
    const float* __restrict__ g,
    const float* __restrict__ lw1, const float* __restrict__ lb1,
    const float* __restrict__ lw2, const float* __restrict__ lb2,
    const float* __restrict__ lw3, const float* __restrict__ lb3,
    float* __restrict__ out)
{
    __shared__ float gs[H1], a1[H1], a2[H2];
    const int t = threadIdx.x;
    if (t < H1) gs[t] = g[t];
    __syncthreads();
    if (t < H1) {
        float s = lb1[t];
        for (int i = 0; i < H1; ++i) s += gs[i] * lw1[i * H1 + t];
        a1[t] = fmaxf(s, 0.0f);
    }
    __syncthreads();
    if (t < H2) {
        float s = lb2[t];
        for (int i = 0; i < H1; ++i) s += a1[i] * lw2[i * H2 + t];
        a2[t] = fmaxf(s, 0.0f);
    }
    __syncthreads();
    if (t == 0) {
        float s = lb3[0];
        for (int i = 0; i < H2; ++i) s += a2[i] * lw3[i];
        out[0] = s;
    }
}

// ---------------------------------------------------------------------------
extern "C" void kernel_launch(void* const* d_in, const int* in_sizes, int n_in,
                              void* d_out, int out_size, void* d_ws, size_t ws_size,
                              hipStream_t stream) {
    const float*     x   = (const float*)d_in[0];
    const long long* ei  = (const long long*)d_in[1];
    const float*     W1  = (const float*)d_in[2];
    const float*     b1  = (const float*)d_in[3];
    const float*     lw1 = (const float*)d_in[4];
    const float*     lb1 = (const float*)d_in[5];
    const float*     lw2 = (const float*)d_in[6];
    const float*     lb2 = (const float*)d_in[7];
    const float*     lw3 = (const float*)d_in[8];
    const float*     lb3 = (const float*)d_in[9];

    const int N = in_sizes[0] / SD;   // 50000
    const int E = in_sizes[1] / 2;    // 800000
    const long long* src = ei;
    const long long* dst = ei + E;

    // workspace: dinv[N] | g[pad] | wsw[12288 bf16] | hs[N*96] | S[N*96]
    char* ws = (char*)d_ws;
    size_t off = 0;
    float*  dinv = (float*)(ws + off);  off += ((size_t)N * 4 + 255) & ~(size_t)255;
    float*  g    = (float*)(ws + off);  off += 256;
    __bf16* wsw  = (__bf16*)(ws + off); off += ((size_t)WSW_ELEMS * 2 + 255) & ~(size_t)255;
    float*  hs   = (float*)(ws + off);  off += ((size_t)N * H1 * 4 + 255) & ~(size_t)255;
    float*  S    = (float*)(ws + off);

    deg_init_kernel  <<<(N + 255) / 256, 256, 0, stream>>>(dinv, g, N);
    deg_count_kernel <<<(E + 255) / 256, 256, 0, stream>>>(dst, dinv, E);
    deg_rsqrt_kernel <<<(N + 255) / 256, 256, 0, stream>>>(dinv, N);
    w_swizzle_kernel <<<(WSW_ELEMS + 255) / 256, 256, 0, stream>>>(W1, wsw);
    gemm_scale_kernel<<<(N + 127) / 128, 256, 0, stream>>>(x, wsw, dinv, hs, S, N);
    edge_scatter_kernel<<<(E + 7) / 8, 256, 0, stream>>>(src, dst, hs, S, E);
    pool_kernel      <<<(N + 511) / 512, 256, 0, stream>>>(S, dinv, b1, g, N);
    mlp_kernel       <<<1, 128, 0, stream>>>(g, lw1, lb1, lw2, lb2, lw3, lb3, (float*)d_out);
}
